// GPSDPAttention_7413113553037
// MI455X (gfx1250) — matrix operand order repaired and checked by reading, not compiled, not run on hardware
//
#include <hip/hip_runtime.h>
#include <hip/hip_bf16.h>

typedef __attribute__((ext_vector_type(16))) _Float16 h16v;
typedef __attribute__((ext_vector_type(8)))  _Float16 h8v;
typedef __attribute__((ext_vector_type(8)))  float    f8v;

#define WMMA_F16(A, B, C) \
    __builtin_amdgcn_wmma_f32_16x16x32_f16(false, (A), false, (B), (short)0, (C), false, false)

// Problem constants (match reference setup_inputs)
#define BATCH 8
#define HEADS 4
#define SEQL  2048
#define DHEAD 128
#define TK    32      // keys per tile
#define WAVES 8       // 256 threads
#define ROWS_PER_BLOCK 128

__launch_bounds__(256, 1)
__global__ void gpsdp_flash_attn(const float* __restrict__ Qg,
                                 const float* __restrict__ Kg,
                                 const float* __restrict__ Vg,
                                 float* __restrict__ Og) {
    // LDS: K tile [key][d] f16 (8KB), V tile transposed [d][key] f16 (8KB),
    // per-wave P staging 16x32 f16 (8KB total)
    __shared__ _Float16 k_lds[TK * DHEAD];
    __shared__ _Float16 v_lds[DHEAD * TK];
    __shared__ _Float16 p_lds[WAVES * 16 * TK];

    const int tid  = threadIdx.x;
    const int wave = tid >> 5;
    const int lane = tid & 31;
    const int lo   = lane & 15;   // column/N coordinate inside fragments
    const int hi   = lane >> 4;   // half-wave selector

    const int h  = blockIdx.y;
    const int b  = blockIdx.z;
    const int q0 = blockIdx.x * ROWS_PER_BLOCK + wave * 16;

    const size_t bh = ((size_t)b * HEADS + h) * (size_t)SEQL * DHEAD;
    const float* Qp = Qg + bh;
    const float* Kp = Kg + bh;
    const float* Vp = Vg + bh;
    float*       Op = Og + bh;

    // sigma_h = 5 * 2^h  ->  1/(2 sigma^2) = 0.02 / 4^h
    const float inv2s2 = 0.02f / (float)(1 << (2 * h));
    const float scale  = 0.08838834764831845f;  // 1/sqrt(128)

    // ---- Load Q A-fragments (scale folded in, f32 -> f16) ----
    // A k order used here (B loads match): row M = lane%16, K = 16*(lane/16) + e
    h16v aQ[4];
    {
        const float* qrow = Qp + (size_t)(q0 + lo) * DHEAD;
#pragma unroll
        for (int c = 0; c < 4; ++c) {
#pragma unroll
            for (int e = 0; e < 16; ++e) {
                const int d = c * 32 + (hi << 4) + e;
                aQ[c][e] = (_Float16)(qrow[d] * scale);
            }
        }
    }

    // ---- online-softmax state ----
    f8v   o[8];
    float m[8], l[8];
#pragma unroll
    for (int n = 0; n < 8; ++n) o[n] = (f8v){};
#pragma unroll
    for (int r = 0; r < 8; ++r) { m[r] = -3.0e38f; l[r] = 0.0f; }

    for (int kt = 0; kt < SEQL; kt += TK) {
        __syncthreads();  // previous iteration's LDS reads are done

        // ---- cooperative K/V tile load: 32x128 f32 each, convert to f16 ----
        const float* Ksrc = Kp + (size_t)kt * DHEAD;
        const float* Vsrc = Vp + (size_t)kt * DHEAD;
        if (kt + TK < SEQL) {
            __builtin_prefetch(Ksrc + TK * DHEAD + tid * 16, 0, 0);
            __builtin_prefetch(Vsrc + TK * DHEAD + tid * 16, 0, 0);
        }
#pragma unroll
        for (int i = 0; i < 4; ++i) {
            const int idx = tid + i * 256;          // float4 index, 0..1023
            const float4 k4 = ((const float4*)Ksrc)[idx];
            const float4 v4 = ((const float4*)Vsrc)[idx];
            const int f   = idx << 2;               // flat element
            const int row = f >> 7;                 // key within tile
            const int col = f & 127;                // d
            k_lds[row * DHEAD + col + 0] = (_Float16)k4.x;
            k_lds[row * DHEAD + col + 1] = (_Float16)k4.y;
            k_lds[row * DHEAD + col + 2] = (_Float16)k4.z;
            k_lds[row * DHEAD + col + 3] = (_Float16)k4.w;
            v_lds[(col + 0) * TK + row] = (_Float16)v4.x;
            v_lds[(col + 1) * TK + row] = (_Float16)v4.y;
            v_lds[(col + 2) * TK + row] = (_Float16)v4.z;
            v_lds[(col + 3) * TK + row] = (_Float16)v4.w;
        }
        __syncthreads();

        // ---- S = (Q*scale) K^T : two 16x16 D-fragments (key halves) ----
        f8v s[2];
        s[0] = (f8v){};
        s[1] = (f8v){};
#pragma unroll
        for (int c = 0; c < 4; ++c) {
#pragma unroll
            for (int kh = 0; kh < 2; ++kh) {
                // B layout: lane column N = lane%16 (key), K(d) = e + 16*(lane/16)
                const h16v bk =
                    *(const h16v*)&k_lds[(kh * 16 + lo) * DHEAD + c * 32 + hi * 16];
                s[kh] = WMMA_F16(aQ[c], bk, s[kh]);
            }
        }

        // ---- additive causal Gaussian bias + online softmax ----
        float alpha[8];
#pragma unroll
        for (int r = 0; r < 8; ++r) {
            const int q = q0 + r + (hi << 3);   // D-frag: M = r + 8*hi
#pragma unroll
            for (int kh = 0; kh < 2; ++kh) {
                const int kcol = kt + kh * 16 + lo;  // D-frag: N = lane%16
                const float diff = (float)(kcol - q);
                const float bias =
                    (q >= kcol) ? __expf(-diff * diff * inv2s2) : 0.0f;
                s[kh][r] += bias;
            }
            float rowm = fmaxf(s[0][r], s[1][r]);
            rowm = fmaxf(rowm, __shfl_xor(rowm, 1));
            rowm = fmaxf(rowm, __shfl_xor(rowm, 2));
            rowm = fmaxf(rowm, __shfl_xor(rowm, 4));
            rowm = fmaxf(rowm, __shfl_xor(rowm, 8));
            const float mnew = fmaxf(m[r], rowm);
            alpha[r] = __expf(m[r] - mnew);
            const float p0 = __expf(s[0][r] - mnew);
            const float p1 = __expf(s[1][r] - mnew);
            s[0][r] = p0;
            s[1][r] = p1;
            float rs = p0 + p1;
            rs += __shfl_xor(rs, 1);
            rs += __shfl_xor(rs, 2);
            rs += __shfl_xor(rs, 4);
            rs += __shfl_xor(rs, 8);
            l[r] = l[r] * alpha[r] + rs;
            m[r] = mnew;
        }
#pragma unroll
        for (int n = 0; n < 8; ++n)
#pragma unroll
            for (int r = 0; r < 8; ++r) o[n][r] *= alpha[r];

        // ---- stage P (16x32 f16) through wave-private LDS to A-frag layout ----
        _Float16* pw = p_lds + wave * (16 * TK);
#pragma unroll
        for (int r = 0; r < 8; ++r) {
            const int M = r + (hi << 3);
            pw[M * TK + lo]      = (_Float16)s[0][r];
            pw[M * TK + 16 + lo] = (_Float16)s[1][r];
        }
        asm volatile("s_wait_dscnt 0" ::: "memory");  // wave-local LDS RAW
        const h8v pa = *(const h8v*)&pw[lo * TK + (hi << 4)];
        const h8v pb = *(const h8v*)&pw[lo * TK + (hi << 4) + 8];
        h16v aP;
#pragma unroll
        for (int e = 0; e < 8; ++e) { aP[e] = pa[e]; aP[e + 8] = pb[e]; }

        // ---- O += P V : 8 N-chunks of 16 over transposed V tile ----
#pragma unroll
        for (int n = 0; n < 8; ++n) {
            const h16v bv = *(const h16v*)&v_lds[(n * 16 + lo) * TK + hi * 16];
            o[n] = WMMA_F16(aP, bv, o[n]);
        }
    }

    // ---- epilogue: normalize and store fp32 output ----
#pragma unroll
    for (int r = 0; r < 8; ++r) {
        const int   q    = q0 + r + (hi << 3);
        const float invl = 1.0f / l[r];
        float* orow = Op + (size_t)q * DHEAD;
#pragma unroll
        for (int n = 0; n < 8; ++n) {
            orow[n * 16 + lo] = o[n][r] * invl;
        }
    }
}

extern "C" void kernel_launch(void* const* d_in, const int* in_sizes, int n_in,
                              void* d_out, int out_size, void* d_ws, size_t ws_size,
                              hipStream_t stream) {
    const float* Q = (const float*)d_in[0];
    const float* K = (const float*)d_in[1];
    const float* V = (const float*)d_in[2];
    // d_in[3] = attn_mask, unused by the reference forward
    float* O = (float*)d_out;

    dim3 grid(SEQL / ROWS_PER_BLOCK, HEADS, BATCH);  // (16, 4, 8)
    dim3 block(256);
    gpsdp_flash_attn<<<grid, block, 0, stream>>>(Q, K, V, O);
}
